// ExploratoryMechanism_87411174408613
// MI455X (gfx1250) — matrix operand order repaired and checked by reading, not compiled, not run on hardware
//
#include <hip/hip_runtime.h>
#include <hip/hip_bf16.h>
#include <stdint.h>

#define BB 4
#define NQ 1024
#define NC 16384
#define DD 1024
#define TOPN 5

typedef __attribute__((ext_vector_type(16))) __bf16 v16bf;
typedef __attribute__((ext_vector_type(8)))  __bf16 v8bf;
typedef __attribute__((ext_vector_type(8)))  float  v8f;

union AFrag { v16bf v; v8bf h[2]; };

__device__ __forceinline__ unsigned short f2bf(float f) {
  union { float f; unsigned u; } x; x.f = f;
  unsigned u = x.u;
  u += 0x7FFFu + ((u >> 16) & 1u);   // round-to-nearest-even
  return (unsigned short)(u >> 16);
}

// ---------------- f32 -> bf16 bulk convert ----------------
__global__ void k_cvt_bf16(const float* __restrict__ src,
                           unsigned short* __restrict__ dst, int n4) {
  int i = blockIdx.x * blockDim.x + threadIdx.x;
  const int stride = gridDim.x * blockDim.x;
  for (; i < n4; i += stride) {
    float4 v = reinterpret_cast<const float4*>(src)[i];
    ushort4 o;
    o.x = f2bf(v.x); o.y = f2bf(v.y); o.z = f2bf(v.z); o.w = f2bf(v.w);
    reinterpret_cast<ushort4*>(dst)[i] = o;
  }
}

// ---------------- row squared-norm ----------------
__global__ void k_rowsq(const float* __restrict__ src, float* __restrict__ dst,
                        int dcols) {
  __shared__ float red[256];
  const int row = blockIdx.x;
  const float* p = src + (size_t)row * dcols;
  float s = 0.f;
  for (int j = threadIdx.x; j < dcols; j += blockDim.x) { float v = p[j]; s += v * v; }
  red[threadIdx.x] = s;
  __syncthreads();
  for (int off = 128; off > 0; off >>= 1) {
    if ((int)threadIdx.x < off) red[threadIdx.x] += red[threadIdx.x + off];
    __syncthreads();
  }
  if (threadIdx.x == 0) dst[row] = red[0];
}

// ---------------- projection GEMM: q_proj = q @ W^T + b ----------------
__global__ __launch_bounds__(128) void k_proj(
    const unsigned short* __restrict__ qb,   // [BB*NQ, DD] bf16
    const unsigned short* __restrict__ wb,   // [DD, DD] bf16 (row e = output dim)
    const float* __restrict__ bias,          // [DD]
    float* __restrict__ pf,                  // [BB*NQ, DD] f32 out
    unsigned short* __restrict__ pb) {       // [BB*NQ, DD] bf16 out
  const int lane = threadIdx.x & 31;
  const int wave = threadIdx.x >> 5;
  const int tile = blockIdx.x * 4 + wave;
  const int ET = DD / 16;                 // 64 output-dim tiles
  const int et = tile % ET;
  const int rt = tile / ET;               // 0 .. BB*NQ/16-1
  const int half = lane >> 4;
  const int m = lane & 15;
  const unsigned short* arow = qb + (size_t)(rt * 16 + m) * DD;
  const unsigned short* brow = wb + (size_t)(et * 16 + m) * DD;
  v8f acc = {};
#pragma unroll 4
  for (int k0 = 0; k0 < DD; k0 += 32) {
    AFrag a;
    a.h[0] = *reinterpret_cast<const v8bf*>(arow + k0 + half * 16);
    a.h[1] = *reinterpret_cast<const v8bf*>(arow + k0 + half * 16 + 8);
    v16bf bv = *reinterpret_cast<const v16bf*>(brow + k0 + half * 16);
    acc = __builtin_amdgcn_wmma_f32_16x16x32_bf16(false, a.v, false, bv,
                                                  (short)0, acc, false, false);
  }
  const float bias_n = bias[et * 16 + m];
#pragma unroll
  for (int r = 0; r < 8; ++r) {
    const int row = rt * 16 + r + 8 * half;   // C layout: M = r + 8*(lane/16)
    const int col = et * 16 + m;              // N = lane%16
    const float v = acc[r] + bias_n;
    pf[(size_t)row * DD + col] = v;
    pb[(size_t)row * DD + col] = f2bf(v);
  }
}

// ---------------- fused distance + top-5 ----------------
// Block = 4 waves, owns 16 query rows (LDS-staged). Each wave processes 64
// context columns (4 N-tiles) per iteration: one A fragment feeds 4 WMMAs.
__global__ __launch_bounds__(128) void k_dist_topk(
    const unsigned short* __restrict__ pb,   // q_proj bf16 [BB*NQ, DD]
    const unsigned short* __restrict__ cb,   // context bf16 [BB*NC, DD]
    const float* __restrict__ qsq,           // [BB*NQ]
    const float* __restrict__ csq,           // [BB*NC]
    float* __restrict__ out_d,               // [BB*NQ*TOPN]
    int* __restrict__ out_i) {               // [BB*NQ*TOPN]
  __shared__ __align__(16) unsigned short ldsA[16 * DD];   // 32 KB query tile
  __shared__ float sD[4][16][68];                          // 17 KB, pad=4 banks
  __shared__ float mgD[4][16][TOPN];
  __shared__ int   mgI[4][16][TOPN];

  const int lane = threadIdx.x & 31;
  const int wave = threadIdx.x >> 5;
  const int half = lane >> 4;
  const int m = lane & 15;
  const int bq = blockIdx.x;              // 0 .. BB*NQ/16-1
  const int b  = bq / (NQ / 16);
  const int qrow0 = bq * 16;              // global query row base

  {
    const uint4* src = reinterpret_cast<const uint4*>(pb + (size_t)qrow0 * DD);
    uint4* dst = reinterpret_cast<uint4*>(ldsA);
    for (int i = threadIdx.x; i < 16 * DD / 8; i += blockDim.x) dst[i] = src[i];
  }
  __syncthreads();

  float td[TOPN]; int ti[TOPN];
#pragma unroll
  for (int i = 0; i < TOPN; ++i) { td[i] = 3.0e38f; ti[i] = 0; }

  const int per_wave = NC / (64 * 4);     // 64 iterations of 64 columns
  for (int t = 0; t < per_wave; ++t) {
    const int cbase = (wave * per_wave + t) * 64;
    const unsigned short* crow = cb + (size_t)(b * NC + cbase + m) * DD;
    const float csq0 = csq[b * NC + cbase + m];
    const float csq1 = csq[b * NC + cbase + 16 + m];
    const float csq2 = csq[b * NC + cbase + 32 + m];
    const float csq3 = csq[b * NC + cbase + 48 + m];

    // prefetch the start of next chunk's rows into cache (global_prefetch_b8)
    if (t + 1 < per_wave) {
      __builtin_prefetch(crow + (size_t)64 * DD, 0, 3);
      __builtin_prefetch(crow + (size_t)80 * DD, 0, 3);
      __builtin_prefetch(crow + (size_t)96 * DD, 0, 3);
      __builtin_prefetch(crow + (size_t)112 * DD, 0, 3);
    }

    v8f acc0 = {}, acc1 = {}, acc2 = {}, acc3 = {};
#pragma unroll 2
    for (int k0 = 0; k0 < DD; k0 += 32) {
      AFrag a;
      a.h[0] = *reinterpret_cast<const v8bf*>(ldsA + m * DD + k0 + half * 16);
      a.h[1] = *reinterpret_cast<const v8bf*>(ldsA + m * DD + k0 + half * 16 + 8);
      const v16bf b0 = *reinterpret_cast<const v16bf*>(crow + k0 + half * 16);
      const v16bf b1 = *reinterpret_cast<const v16bf*>(crow + (size_t)16 * DD + k0 + half * 16);
      const v16bf b2 = *reinterpret_cast<const v16bf*>(crow + (size_t)32 * DD + k0 + half * 16);
      const v16bf b3 = *reinterpret_cast<const v16bf*>(crow + (size_t)48 * DD + k0 + half * 16);
      acc0 = __builtin_amdgcn_wmma_f32_16x16x32_bf16(false, a.v, false, b0,
                                                     (short)0, acc0, false, false);
      acc1 = __builtin_amdgcn_wmma_f32_16x16x32_bf16(false, a.v, false, b1,
                                                     (short)0, acc1, false, false);
      acc2 = __builtin_amdgcn_wmma_f32_16x16x32_bf16(false, a.v, false, b2,
                                                     (short)0, acc2, false, false);
      acc3 = __builtin_amdgcn_wmma_f32_16x16x32_bf16(false, a.v, false, b3,
                                                     (short)0, acc3, false, false);
    }
#pragma unroll
    for (int r = 0; r < 8; ++r) {
      const int row = r + 8 * half;
      sD[wave][row][m]      = csq0 - 2.0f * acc0[r];
      sD[wave][row][16 + m] = csq1 - 2.0f * acc1[r];
      sD[wave][row][32 + m] = csq2 - 2.0f * acc2[r];
      sD[wave][row][48 + m] = csq3 - 2.0f * acc3[r];
    }
    __syncthreads();
    if (lane < 16) {
      for (int j = 0; j < 64; ++j) {
        const float v = sD[wave][lane][j];
        if (v < td[TOPN - 1]) {
          int p = TOPN - 1;
          while (p > 0 && td[p - 1] > v) { td[p] = td[p - 1]; ti[p] = ti[p - 1]; --p; }
          td[p] = v; ti[p] = cbase + j;
        }
      }
    }
    __syncthreads();
  }

  if (lane < 16) {
#pragma unroll
    for (int i = 0; i < TOPN; ++i) { mgD[wave][lane][i] = td[i]; mgI[wave][lane][i] = ti[i]; }
  }
  __syncthreads();

  if (wave == 0 && lane < 16) {
    float fd[TOPN]; int fi[TOPN];
#pragma unroll
    for (int i = 0; i < TOPN; ++i) { fd[i] = 3.0e38f; fi[i] = 0; }
    for (int w = 0; w < 4; ++w)
      for (int i = 0; i < TOPN; ++i) {
        const float v = mgD[w][lane][i]; const int idx = mgI[w][lane][i];
        if (v < fd[TOPN - 1]) {
          int p = TOPN - 1;
          while (p > 0 && (fd[p - 1] > v || (fd[p - 1] == v && fi[p - 1] > idx))) {
            fd[p] = fd[p - 1]; fi[p] = fi[p - 1]; --p;
          }
          fd[p] = v; fi[p] = idx;
        }
      }
    const int qrow = qrow0 + lane;
    const float qs = qsq[qrow];
    for (int i = 0; i < TOPN; ++i) {
      float d2 = qs + fd[i];
      d2 = d2 > 0.f ? d2 : 0.f;
      out_d[qrow * TOPN + i] = sqrtf(d2);
      out_i[qrow * TOPN + i] = fi[i];
    }
  }
}

extern "C" void kernel_launch(void* const* d_in, const int* in_sizes, int n_in,
                              void* d_out, int out_size, void* d_ws, size_t ws_size,
                              hipStream_t stream) {
  const float* q    = (const float*)d_in[0];   // [BB, NQ, DD]
  const float* c    = (const float*)d_in[1];   // [BB, NC, DD]
  const float* W    = (const float*)d_in[2];   // [DD, DD]
  const float* bias = (const float*)d_in[3];   // [DD]
  (void)in_sizes; (void)n_in; (void)out_size; (void)ws_size;

  char* ws = (char*)d_ws;
  size_t off = 0;
  auto alloc = [&](size_t bytes) -> void* {
    void* p = ws + off;
    off = (off + bytes + 255) & ~(size_t)255;
    return p;
  };
  unsigned short* q_bf  = (unsigned short*)alloc((size_t)BB * NQ * DD * 2);
  unsigned short* w_bf  = (unsigned short*)alloc((size_t)DD * DD * 2);
  unsigned short* c_bf  = (unsigned short*)alloc((size_t)BB * NC * DD * 2);
  float*          p_f32 = (float*)alloc((size_t)BB * NQ * DD * 4);
  unsigned short* p_bf  = (unsigned short*)alloc((size_t)BB * NQ * DD * 2);
  float*          qsq   = (float*)alloc((size_t)BB * NQ * 4);
  float*          csq   = (float*)alloc((size_t)BB * NC * 4);

  k_cvt_bf16<<<2048, 256, 0, stream>>>(q, q_bf, BB * NQ * DD / 4);
  k_cvt_bf16<<<1024, 256, 0, stream>>>(W, w_bf, DD * DD / 4);
  k_cvt_bf16<<<8192, 256, 0, stream>>>(c, c_bf, BB * NC * DD / 4);
  k_rowsq<<<BB * NC, 256, 0, stream>>>(c, csq, DD);

  // 16384 output tiles, 4 waves per 128-thread block -> 4096 blocks
  k_proj<<<BB * (NQ / 16) * (DD / 16) / 4, 128, 0, stream>>>(q_bf, w_bf, bias,
                                                             p_f32, p_bf);
  k_rowsq<<<BB * NQ, 256, 0, stream>>>(p_f32, qsq, DD);

  float* out_d = (float*)d_out;
  int*   out_i = (int*)d_out + (size_t)BB * NQ * TOPN;
  k_dist_topk<<<BB * NQ / 16, 128, 0, stream>>>(p_bf, c_bf, qsq, csq, out_d, out_i);
}
